// EdgeUpdateBlock_60120952209605
// MI455X (gfx1250) — matrix operand order repaired and checked by reading, not compiled, not run on hardware
//
#include <hip/hip_runtime.h>
#include <hip/hip_bf16.h>

typedef __attribute__((ext_vector_type(8)))  float          v8f;
typedef __attribute__((ext_vector_type(4)))  float          v4f;
typedef __attribute__((ext_vector_type(16))) __bf16         v16bf;
typedef __attribute__((ext_vector_type(4)))  unsigned short u16x4;
typedef __attribute__((ext_vector_type(8)))  unsigned short u16x8;
typedef __attribute__((ext_vector_type(16))) unsigned short u16x16;

// ---------------------------------------------------------------------------
// helpers
// ---------------------------------------------------------------------------
// Native f32 -> bf16 convert (lowers to v_cvt_pk_bf16_f32 on gfx1250).
__device__ __forceinline__ unsigned short f2bf(float f) {
    return __builtin_bit_cast(unsigned short, (__bf16)f);
}

__device__ __forceinline__ v16bf combine8(u16x8 lo, u16x8 hi) {
    u16x16 r;
#pragma unroll
    for (int i = 0; i < 8; ++i) { r[i] = lo[i]; r[i + 8] = hi[i]; }
    return __builtin_bit_cast(v16bf, r);
}

// A fragment: 16x32 bf16 tile, row-major in LDS with row stride `rs` ushorts.
// ISA layout: lanes 0-15 (M=lane): K 0-7 then 16-23 ; lanes 16-31 (M=lane-16): K 8-15 then 24-31.
__device__ __forceinline__ v16bf load_a_frag(const unsigned short* base, int rs,
                                             int kbase, int l15, int lg) {
    const unsigned short* p = base + l15 * rs + kbase + 8 * lg;
    return combine8(*(const u16x8*)p, *(const u16x8*)(p + 16));
}

// B fragment: stored fragment-major: tile = 512 ushorts, lane's 16 values contiguous.
__device__ __forceinline__ v16bf load_b_frag(const unsigned short* tile, int lane) {
    const unsigned short* p = tile + lane * 16;
    return combine8(*(const u16x8*)p, *(const u16x8*)(p + 8));
}

__device__ __forceinline__ v8f wmma_bf16(v16bf a, v16bf b, v8f c) {
    return __builtin_amdgcn_wmma_f32_16x16x32_bf16(false, a, false, b,
                                                   (short)0, c, false, false);
}

__device__ __forceinline__ void lds_fence() {
    asm volatile("s_wait_dscnt 0" ::: "memory");
}

// Pack a KxN f32 weight matrix (row stride = ncols) into B-fragment layout:
// frag element j of lane in tile (kb,nb): K = kb*32 + 16*(j>>3) + 8*(lane>>4) + (j&7), N = nb*16 + (lane&15)
__device__ void pack_weight(unsigned short* dst, const float* W, int ncols,
                            int nbCnt, int kbCnt, int tid, int nthreads) {
    int total = kbCnt * nbCnt * 512;
    for (int idx = tid; idx < total; idx += nthreads) {
        int tile = idx >> 9, rem = idx & 511;
        int ln = rem >> 4, j = rem & 15;
        int kb = tile / nbCnt, nb = tile - kb * nbCnt;
        int k = kb * 32 + ((j >> 3) << 4) + ((ln >> 4) << 3) + (j & 7);
        int n = nb * 16 + (ln & 15);
        dst[idx] = f2bf(W[(size_t)k * ncols + n]);
    }
}

// ---------------------------------------------------------------------------
// workspace zeroing (ws is poisoned by harness; must re-zero every launch)
// ---------------------------------------------------------------------------
__global__ void zero_kernel(float* __restrict__ p, long n) {
    long i = (long)blockIdx.x * blockDim.x + threadIdx.x;
    long st = (long)gridDim.x * blockDim.x;
    for (; i < n; i += st) p[i] = 0.0f;
}

// ---------------------------------------------------------------------------
// Edge kernel: 4 waves / block. Each wave owns a 16-edge tile at a time.
// LDS (ushorts): Wmsg frags 4096 | We1 frags 24576 | We2 frags 8192 |
//                per-wave Ain 16x192 (x4) | per-wave H 16x128 (x4)
// ---------------------------------------------------------------------------
#define EDGE_LDS_USHORTS (4096 + 24576 + 8192 + 4 * 3072 + 4 * 2048)

__global__ void edge_kernel(const float* __restrict__ x,
                            const int* __restrict__ srcI,
                            const int* __restrict__ dstI,
                            const float* __restrict__ e,
                            const float* __restrict__ Wmsg,
                            const float* __restrict__ We1,
                            const float* __restrict__ be1,
                            const float* __restrict__ We2,
                            const float* __restrict__ be2,
                            float* __restrict__ eout,
                            float* __restrict__ sums,
                            float* __restrict__ cnt,
                            int ntiles) {
    extern __shared__ unsigned short smem[];
    unsigned short* wmsgF = smem;               // 2kb x 4nb
    unsigned short* we1F  = wmsgF + 4096;       // 6kb x 8nb
    unsigned short* we2F  = we1F + 24576;       // 4kb x 4nb
    const int wave = threadIdx.x >> 5;
    const int lane = threadIdx.x & 31;
    const int l15 = lane & 15, lg = lane >> 4;
    unsigned short* ain = we2F + 8192 + wave * 3072;              // 16 x 192
    unsigned short* hst = we2F + 8192 + 4 * 3072 + wave * 2048;   // 16 x 128

    pack_weight(wmsgF, Wmsg, 64, 4, 2, threadIdx.x, blockDim.x);
    pack_weight(we1F,  We1, 128, 8, 6, threadIdx.x, blockDim.x);
    pack_weight(we2F,  We2,  64, 4, 4, threadIdx.x, blockDim.x);
    __syncthreads();

    // loop-invariant bias values (per-lane)
    float bv1r[8], bv2r[4];
#pragma unroll
    for (int nb = 0; nb < 8; ++nb) bv1r[nb] = be1[nb * 16 + l15];
#pragma unroll
    for (int nb = 0; nb < 4; ++nb) bv2r[nb] = be2[nb * 16 + l15];

    const int gwave = blockIdx.x * (blockDim.x >> 5) + wave;
    const int nwaves = gridDim.x * (blockDim.x >> 5);

    for (int t = gwave; t < ntiles; t += nwaves) {
        const int base = t * 16;

        // ---- stage xs (cols 0..63) and xd (cols 64..127) as bf16 ----
        {
            const int row = l15;
            const int sR = srcI[base + row];
            const int dR = dstI[base + row];
            const float* xsrow = x + (size_t)sR * 64 + lg * 32;
            const float* xdrow = x + (size_t)dR * 64 + lg * 32;
#pragma unroll
            for (int i = 0; i < 32; i += 4) {
                v4f vs = *(const v4f*)(xsrow + i);
                v4f vd = *(const v4f*)(xdrow + i);
                u16x4 hs = { f2bf(vs[0]), f2bf(vs[1]), f2bf(vs[2]), f2bf(vs[3]) };
                u16x4 hd = { f2bf(vd[0]), f2bf(vd[1]), f2bf(vd[2]), f2bf(vd[3]) };
                *(u16x4*)&ain[row * 192 + lg * 32 + i]      = hs;
                *(u16x4*)&ain[row * 192 + 64 + lg * 32 + i] = hd;
            }
        }
        lds_fence();

        // destination node indices for this tile's 8 rows owned by this lane
        int didx[8];
#pragma unroll
        for (int r = 0; r < 8; ++r) didx[r] = dstI[base + r + 8 * lg];

        // ---- m = xs @ Wmsg ; cols 128..191 = bf16(e + m) ----
        {
            // A fragments over xs (kb = 0,1) are shared by all 4 nb tiles
            v16bf a0 = load_a_frag(ain, 192, 0, l15, lg);
            v16bf a1 = load_a_frag(ain, 192, 32, l15, lg);
            for (int nb = 0; nb < 4; ++nb) {
                v8f c = {};
                c = wmma_bf16(a0, load_b_frag(wmsgF + (0 * 4 + nb) * 512, lane), c);
                c = wmma_bf16(a1, load_b_frag(wmsgF + (1 * 4 + nb) * 512, lane), c);
                const int col = nb * 16 + l15;
#pragma unroll
                for (int r = 0; r < 8; ++r) {
                    const int m_ = r + 8 * lg;
                    float ev = e[(size_t)(base + m_) * 64 + col];
                    ain[m_ * 192 + 128 + col] = f2bf(ev + c[r]);
                }
            }
        }
        lds_fence();

        // ---- h = relu(e_in @ We1 + b1) ----
        for (int nb = 0; nb < 8; ++nb) {
            const float bv = bv1r[nb];
            v8f c = { bv, bv, bv, bv, bv, bv, bv, bv };
#pragma unroll
            for (int kb = 0; kb < 6; ++kb) {
                v16bf a = load_a_frag(ain, 192, kb * 32, l15, lg);
                v16bf b = load_b_frag(we1F + (kb * 8 + nb) * 512, lane);
                c = wmma_bf16(a, b, c);
            }
            const int col = nb * 16 + l15;
#pragma unroll
            for (int r = 0; r < 8; ++r)
                hst[(r + 8 * lg) * 128 + col] = f2bf(fmaxf(c[r], 0.0f));
        }
        lds_fence();

        // ---- e_new = e + h @ We2 + b2 ; write + scatter ----
        for (int nb = 0; nb < 4; ++nb) {
            const float bv = bv2r[nb];
            v8f c = { bv, bv, bv, bv, bv, bv, bv, bv };
#pragma unroll
            for (int kb = 0; kb < 4; ++kb) {
                v16bf a = load_a_frag(hst, 128, kb * 32, l15, lg);
                v16bf b = load_b_frag(we2F + (kb * 4 + nb) * 512, lane);
                c = wmma_bf16(a, b, c);
            }
            const int col = nb * 16 + l15;
#pragma unroll
            for (int r = 0; r < 8; ++r) {
                const int m_ = r + 8 * lg;
                const size_t ei = (size_t)(base + m_) * 64 + col;
                float val = e[ei] + c[r];
                eout[ei] = val;
                atomicAdd(&sums[(size_t)didx[r] * 64 + col], val);
            }
        }
        if (lane < 16) atomicAdd(&cnt[dstI[base + lane]], 1.0f);
        lds_fence();
    }
}

// ---------------------------------------------------------------------------
// Node kernel: x_new = x + relu([x | m_dst] @ Wv1 + b1) @ Wv2 + b2
// LDS (ushorts): Wv1 frags 16384 | Wv2 frags 8192 | per-wave Ain 16x128 x4 | H 16x128 x4
// ---------------------------------------------------------------------------
#define NODE_LDS_USHORTS (16384 + 8192 + 4 * 2048 + 4 * 2048)

__global__ void node_kernel(const float* __restrict__ x,
                            const float* __restrict__ Wv1,
                            const float* __restrict__ bv1,
                            const float* __restrict__ Wv2,
                            const float* __restrict__ bv2,
                            const float* __restrict__ sums,
                            const float* __restrict__ cnt,
                            float* __restrict__ xout,
                            int ntiles) {
    extern __shared__ unsigned short smem[];
    unsigned short* wv1F = smem;           // 4kb x 8nb
    unsigned short* wv2F = wv1F + 16384;   // 4kb x 4nb
    const int wave = threadIdx.x >> 5;
    const int lane = threadIdx.x & 31;
    const int l15 = lane & 15, lg = lane >> 4;
    unsigned short* ain = wv2F + 8192 + wave * 2048;              // 16 x 128
    unsigned short* hst = wv2F + 8192 + 4 * 2048 + wave * 2048;   // 16 x 128

    pack_weight(wv1F, Wv1, 128, 8, 4, threadIdx.x, blockDim.x);
    pack_weight(wv2F, Wv2,  64, 4, 4, threadIdx.x, blockDim.x);
    __syncthreads();

    // loop-invariant bias values (per-lane)
    float bv1r[8], bv2r[4];
#pragma unroll
    for (int nb = 0; nb < 8; ++nb) bv1r[nb] = bv1[nb * 16 + l15];
#pragma unroll
    for (int nb = 0; nb < 4; ++nb) bv2r[nb] = bv2[nb * 16 + l15];

    const int gwave = blockIdx.x * (blockDim.x >> 5) + wave;
    const int nwaves = gridDim.x * (blockDim.x >> 5);

    for (int t = gwave; t < ntiles; t += nwaves) {
        const int base = t * 16;

        // ---- stage v_in = [x | sums/max(cnt,1)] as bf16 : lane covers 64 cols ----
        {
            const int row = l15;
            const int node = base + row;
            const float inv = 1.0f / fmaxf(cnt[node], 1.0f);
            const float scale = (lg == 0) ? 1.0f : inv;
            const float* srcp = (lg == 0) ? (x + (size_t)node * 64)
                                          : (sums + (size_t)node * 64);
#pragma unroll
            for (int i = 0; i < 64; i += 4) {
                v4f v = *(const v4f*)(srcp + i);
                u16x4 h = { f2bf(v[0] * scale), f2bf(v[1] * scale),
                            f2bf(v[2] * scale), f2bf(v[3] * scale) };
                *(u16x4*)&ain[row * 128 + lg * 64 + i] = h;
            }
        }
        lds_fence();

        // ---- h = relu(v_in @ Wv1 + b1) ----
        for (int nb = 0; nb < 8; ++nb) {
            const float bv = bv1r[nb];
            v8f c = { bv, bv, bv, bv, bv, bv, bv, bv };
#pragma unroll
            for (int kb = 0; kb < 4; ++kb) {
                v16bf a = load_a_frag(ain, 128, kb * 32, l15, lg);
                v16bf b = load_b_frag(wv1F + (kb * 8 + nb) * 512, lane);
                c = wmma_bf16(a, b, c);
            }
            const int col = nb * 16 + l15;
#pragma unroll
            for (int r = 0; r < 8; ++r)
                hst[(r + 8 * lg) * 128 + col] = f2bf(fmaxf(c[r], 0.0f));
        }
        lds_fence();

        // ---- x_new = x + h @ Wv2 + b2 ----
        for (int nb = 0; nb < 4; ++nb) {
            const float bv = bv2r[nb];
            v8f c = { bv, bv, bv, bv, bv, bv, bv, bv };
#pragma unroll
            for (int kb = 0; kb < 4; ++kb) {
                v16bf a = load_a_frag(hst, 128, kb * 32, l15, lg);
                v16bf b = load_b_frag(wv2F + (kb * 4 + nb) * 512, lane);
                c = wmma_bf16(a, b, c);
            }
            const int col = nb * 16 + l15;
#pragma unroll
            for (int r = 0; r < 8; ++r) {
                const int m_ = r + 8 * lg;
                const size_t xi = (size_t)(base + m_) * 64 + col;
                xout[xi] = x[xi] + c[r];
            }
        }
        lds_fence();
    }
}

// ---------------------------------------------------------------------------
// launch
// ---------------------------------------------------------------------------
extern "C" void kernel_launch(void* const* d_in, const int* in_sizes, int n_in,
                              void* d_out, int out_size, void* d_ws, size_t ws_size,
                              hipStream_t stream) {
    const float* x      = (const float*)d_in[0];
    const int*   eidx   = (const int*)d_in[1];   // [2,E]: row0 = src, row1 = dst
    const float* e      = (const float*)d_in[2];
    const float* W_msg  = (const float*)d_in[3];
    const float* W_e1   = (const float*)d_in[4];
    const float* b_e1   = (const float*)d_in[5];
    const float* W_e2   = (const float*)d_in[6];
    const float* b_e2   = (const float*)d_in[7];
    const float* W_v1   = (const float*)d_in[8];
    const float* b_v1   = (const float*)d_in[9];
    const float* W_v2   = (const float*)d_in[10];
    const float* b_v2   = (const float*)d_in[11];

    const int Nn = in_sizes[0] / 64;   // 50000
    const int Ee = in_sizes[2] / 64;   // 800000
    const int* srcI = eidx;
    const int* dstI = eidx + Ee;

    float* xout = (float*)d_out;                   // [N,64]
    float* eout = (float*)d_out + (size_t)Nn * 64; // [E,64]
    float* sums = (float*)d_ws;                    // [N,64]
    float* cntp = sums + (size_t)Nn * 64;          // [N]

    // 1) zero scatter accumulators
    zero_kernel<<<2048, 256, 0, stream>>>(sums, (long)Nn * 65);

    // 2) edge MLP + scatter (16 edges per wave-tile; E is a multiple of 16)
    const int etiles = Ee / 16;
    edge_kernel<<<800, 128, EDGE_LDS_USHORTS * 2, stream>>>(
        x, srcI, dstI, e, W_msg, W_e1, b_e1, W_e2, b_e2,
        eout, sums, cntp, etiles);

    // 3) node MLP (16 nodes per wave-tile; N is a multiple of 16)
    const int ntiles_n = Nn / 16;
    node_kernel<<<200, 128, NODE_LDS_USHORTS * 2, stream>>>(
        x, W_v1, b_v1, W_v2, b_v2, sums, cntp, xout, ntiles_n);
}
